// CALoraLinear_3135326126145
// MI455X (gfx1250) — compile-verified, matrix operand order checked by reading, not run on hardware
//
#include <hip/hip_runtime.h>
#include <stdint.h>

// ---------------- problem constants ----------------
#define B_SZ    512
#define IN_DIM  4096
#define OUT_DIM 4096
#define NCLS    64
#define TRCOLS  512        // NUM_CLASS * RANK (lora_B second dim)
#define LORA_SCALE 16.0f   // SCALING(=2) * RANK-repetition factor(=8)

// ---------------- WMMA types (CDNA5 / gfx1250, wave32) ----------------
typedef __attribute__((ext_vector_type(16))) __bf16 v16bf;
typedef __attribute__((ext_vector_type(2)))  __bf16 v2bf;
typedef __attribute__((ext_vector_type(8)))  float  v8f;

union FragAB {
    uint4 q[2];   // two 16-byte LDS loads
    v16bf v;      // 16 bf16 per lane = 8 VGPRs (16x32 bf16 fragment)
};

// two f32 -> packed bf16 pair (lo in [15:0]).
#if __has_builtin(__builtin_amdgcn_cvt_pk_bf16_f32)
__device__ __forceinline__ uint32_t pack_bf16(float lo, float hi) {
    union { v2bf v; uint32_t u; } cvt;
    cvt.v = __builtin_amdgcn_cvt_pk_bf16_f32(lo, hi);
    return cvt.u;
}
#else
// round-half-up (+0x8000) then single V_PERM_B32 to pack both high halves:
// dst bytes {b3,b2,a3,a2}  (3 VALU total vs ~8 for exact-RNE bit twiddling)
__device__ __forceinline__ uint32_t pack_bf16(float lo, float hi) {
    uint32_t a = __float_as_uint(lo) + 0x8000u;
    uint32_t b = __float_as_uint(hi) + 0x8000u;
    return __builtin_amdgcn_perm(b, a, 0x07060302u);
}
#endif

// ---------------- kernel 1: per-sample top-2 class ids ----------------
__global__ void topk_kernel(const float* __restrict__ pidx, int* __restrict__ cls) {
    int b = blockIdx.x * blockDim.x + threadIdx.x;
    if (b >= B_SZ) return;
    const float* p = pidx + b * NCLS;
    float best = -INFINITY; int bi = 0;
    for (int i = 0; i < NCLS; ++i) { float v = p[i]; if (v > best) { best = v; bi = i; } }
    float best2 = -INFINITY; int bi2 = 0;
    for (int i = 0; i < NCLS; ++i) {
        if (i == bi) continue;
        float v = p[i]; if (v > best2) { best2 = v; bi2 = i; }
    }
    cls[2 * b + 0] = bi;
    cls[2 * b + 1] = bi2;
}

// ---------------- kernel 2: h[b,k] = dot(lora_A[c_k], x[b]) ----------------
__global__ __launch_bounds__(128) void hvec_kernel(
    const float* __restrict__ x, const float* __restrict__ loraA,
    const int* __restrict__ cls, float* __restrict__ h)
{
    const int b = blockIdx.x;
    const int k = blockIdx.y;
    const int c = cls[2 * b + k];                 // faithful: class index row, no rank offset
    const float4* xr = (const float4*)(x + (size_t)b * IN_DIM);
    const float4* ar = (const float4*)(loraA + (size_t)c * IN_DIM);
    float acc = 0.f;
    for (int i = threadIdx.x; i < IN_DIM / 4; i += 128) {
        float4 xv = xr[i], av = ar[i];
        acc += xv.x * av.x + xv.y * av.y + xv.z * av.z + xv.w * av.w;
    }
    __shared__ float red[128];
    red[threadIdx.x] = acc;
    __syncthreads();
    for (int s = 64; s > 0; s >>= 1) {
        if (threadIdx.x < s) red[threadIdx.x] += red[threadIdx.x + s];
        __syncthreads();
    }
    if (threadIdx.x == 0) h[2 * b + k] = red[0];
}

// ---------------- kernel 3: bf16-WMMA GEMM + fused LoRA/bias epilogue ----------------
// Block tile 128x128, BK=32, 8 wave32s (wave tile 64x32 = 4x2 fragments),
// double-buffered LDS: global loads for tile kt+1 in flight during WMMAs on tile kt.
#define BM 128
#define BN 128
#define BK 32
#define NT (IN_DIM / BK)          // 128 K-tiles
#define KPITCH (BK / 2)           // 16 dwords of packed bf16 per row
#define KPAD   (KPITCH + 4)       // padded to 20 dwords (80 B, 16B-aligned) vs bank conflicts

__global__ __launch_bounds__(256) void gemm_lora_kernel(
    const float* __restrict__ x, const float* __restrict__ w,
    const float* __restrict__ bias, const float* __restrict__ loraB,
    const int* __restrict__ cls, const float* __restrict__ hvec,
    float* __restrict__ out)
{
    __shared__ uint32_t sA[2][BM * KPAD];   // 2 x 10 KB bf16 tiles of x
    __shared__ uint32_t sB[2][BN * KPAD];   // 2 x 10 KB bf16 tiles of weight
    __shared__ int      s_c[2 * BM];
    __shared__ float    s_h[2 * BM];

    const int tid   = threadIdx.x;
    const int lane  = tid & 31;            // wave32
    const int wave  = tid >> 5;            // 8 waves
    const int waveM = wave >> 2;           // 0..1  -> 64-row strip
    const int waveN = wave & 3;            // 0..3  -> 32-col strip
    const int m0 = blockIdx.y * BM;
    const int n0 = blockIdx.x * BN;

    // per-sample LoRA params for this M-tile
    if (tid < BM) {
        s_c[2 * tid]     = cls[2 * (m0 + tid)];
        s_c[2 * tid + 1] = cls[2 * (m0 + tid) + 1];
        s_h[2 * tid]     = hvec[2 * (m0 + tid)];
        s_h[2 * tid + 1] = hvec[2 * (m0 + tid) + 1];
    }

    v8f acc[4][2] = {};   // 64 fp32 accumulator VGPRs per lane

    // staging decomposition: 1024 float4 per tile / 256 threads = 4 each
    const int srow = tid >> 3;             // 0..31 (8 float4 per 32-elem row)
    const int sc4  = tid & 7;
    const float* gA = x + (size_t)m0 * IN_DIM + (size_t)srow * IN_DIM + sc4 * 4;
    const float* gB = w + (size_t)n0 * IN_DIM + (size_t)srow * IN_DIM + sc4 * 4;

    float4 ra[4], rb[4];

    // ---- prologue: stage tile 0 into buffer 0 ----
    #pragma unroll
    for (int i = 0; i < 4; ++i) {
        ra[i] = *(const float4*)(gA + (size_t)(i * 32) * IN_DIM);
        rb[i] = *(const float4*)(gB + (size_t)(i * 32) * IN_DIM);
    }
    #pragma unroll
    for (int i = 0; i < 4; ++i) {
        uint32_t* pa = &sA[0][(i * 32 + srow) * KPAD + sc4 * 2];
        uint32_t* pb = &sB[0][(i * 32 + srow) * KPAD + sc4 * 2];
        pa[0] = pack_bf16(ra[i].x, ra[i].y); pa[1] = pack_bf16(ra[i].z, ra[i].w);
        pb[0] = pack_bf16(rb[i].x, rb[i].y); pb[1] = pack_bf16(rb[i].z, rb[i].w);
    }
    __syncthreads();

    // fragment source column: lanes 0-15 hold K[0..15] of their row,
    // lanes 16-31 hold K[16..31] (contiguous k order, same for A and B); dword offset:
    const int kl = (lane >> 4) * 8;

    // ---- main pipeline ----
    for (int kt = 0; kt < NT; ++kt) {
        const int cur = kt & 1;
        const int nxt = cur ^ 1;
        const bool have_next = (kt + 1) < NT;

        // 1) issue next tile's global loads early (overlap with WMMA below)
        if (have_next) {
            const int k0 = (kt + 1) * BK;
            #pragma unroll
            for (int i = 0; i < 4; ++i) {
                ra[i] = *(const float4*)(gA + (size_t)(i * 32) * IN_DIM + k0);
                rb[i] = *(const float4*)(gB + (size_t)(i * 32) * IN_DIM + k0);
            }
            if (kt + 2 < NT) {   // near-cache hint for tile kt+2 (global_prefetch_b8)
                __builtin_prefetch(gA + (kt + 2) * BK, 0, 3);
                __builtin_prefetch(gB + (kt + 2) * BK, 0, 3);
            }
        }

        // 2) compute on current buffer: 4x2 fragments, 8 WMMAs
        FragAB a[4], b[2];
        #pragma unroll
        for (int fm = 0; fm < 4; ++fm) {
            const uint32_t* p = &sA[cur][(waveM * 64 + fm * 16 + (lane & 15)) * KPAD + kl];
            a[fm].q[0] = *(const uint4*)p;
            a[fm].q[1] = *(const uint4*)(p + 4);
        }
        #pragma unroll
        for (int fn = 0; fn < 2; ++fn) {
            const uint32_t* p = &sB[cur][(waveN * 32 + fn * 16 + (lane & 15)) * KPAD + kl];
            b[fn].q[0] = *(const uint4*)p;
            b[fn].q[1] = *(const uint4*)(p + 4);
        }
        #pragma unroll
        for (int fm = 0; fm < 4; ++fm)
            #pragma unroll
            for (int fn = 0; fn < 2; ++fn)
                acc[fm][fn] = __builtin_amdgcn_wmma_f32_16x16x32_bf16(
                    false, a[fm].v, false, b[fn].v,
                    (short)0, acc[fm][fn], false, false);

        // 3) pack (v_perm_b32) and store next tile into the other buffer
        if (have_next) {
            #pragma unroll
            for (int i = 0; i < 4; ++i) {
                uint32_t* pa = &sA[nxt][(i * 32 + srow) * KPAD + sc4 * 2];
                uint32_t* pb = &sB[nxt][(i * 32 + srow) * KPAD + sc4 * 2];
                pa[0] = pack_bf16(ra[i].x, ra[i].y); pa[1] = pack_bf16(ra[i].z, ra[i].w);
                pb[0] = pack_bf16(rb[i].x, rb[i].y); pb[1] = pack_bf16(rb[i].z, rb[i].w);
            }
        }

        // 4) single barrier per iteration
        __syncthreads();
    }

    // epilogue: out = acc + 2*bias + 16*(loraB[:,c0]*h0 + loraB[:,c1]*h1)
    // C/D layout: VGPR r, lanes 0-15 -> M=r, lanes 16-31 -> M=r+8; N = lane&15
    const int rhalf = (lane >> 4) * 8;
    const int coff  = lane & 15;
    #pragma unroll
    for (int fm = 0; fm < 4; ++fm) {
        #pragma unroll
        for (int fn = 0; fn < 2; ++fn) {
            const int n = n0 + waveN * 32 + fn * 16 + coff;
            const float bs = 2.0f * bias[n];
            const float* lbrow = loraB + (size_t)n * TRCOLS;
            #pragma unroll
            for (int vr = 0; vr < 8; ++vr) {
                const int mloc = waveM * 64 + fm * 16 + vr + rhalf;
                const int m = m0 + mloc;
                const int   c0v = s_c[2 * mloc],     c1v = s_c[2 * mloc + 1];
                const float h0v = s_h[2 * mloc],     h1v = s_h[2 * mloc + 1];
                out[(size_t)m * OUT_DIM + n] =
                    acc[fm][fn][vr] + bs +
                    LORA_SCALE * (lbrow[c0v] * h0v + lbrow[c1v] * h1v);
            }
        }
    }
}

// ---------------- host-side launcher ----------------
extern "C" void kernel_launch(void* const* d_in, const int* in_sizes, int n_in,
                              void* d_out, int out_size, void* d_ws, size_t ws_size,
                              hipStream_t stream) {
    const float* x      = (const float*)d_in[0];
    const float* pidx   = (const float*)d_in[1];
    const float* weight = (const float*)d_in[2];
    const float* bias   = (const float*)d_in[3];
    const float* loraA  = (const float*)d_in[4];
    const float* loraB  = (const float*)d_in[5];
    float* out = (float*)d_out;

    int*   cls = (int*)d_ws;                                     // 2*B ints
    float* h   = (float*)((char*)d_ws + 2 * B_SZ * sizeof(int)); // 2*B floats

    topk_kernel<<<(B_SZ + 255) / 256, 256, 0, stream>>>(pidx, cls);
    hvec_kernel<<<dim3(B_SZ, 2), 128, 0, stream>>>(x, loraA, cls, h);
    gemm_lora_kernel<<<dim3(OUT_DIM / BN, B_SZ / BM), 256, 0, stream>>>(
        x, weight, bias, loraB, cls, h, out);
}
